// GCN_28398323761180
// MI455X (gfx1250) — compile-verified, matrix operand order checked by reading, not run on hardware
//
#include <hip/hip_runtime.h>
#include <hip/hip_bf16.h>

// ---------------------------------------------------------------------------
// 2-layer GCN for MI455X (gfx1250, wave32, WMMA bf16)
//   layer:  h = X@W ;  agg = scatter_add(h[src]*dinv[src]*dinv[dst] -> dst)
//           out = agg + h*dinv^2 + b ; relu (layer 1 only)
// GEMMs use v_wmma_f32_16x16x32_bf16 with fragment-major LDS staging so each
// operand fragment is one 32B contiguous LDS read (2x ds_load_b128).
// ---------------------------------------------------------------------------

#define IN_CH 512
#define HID   128
#define NCLS  64

typedef __attribute__((ext_vector_type(16))) __bf16 v16bf;
typedef __attribute__((ext_vector_type(8)))  float  v8f;

__device__ __forceinline__ unsigned short f2bf_bits(float f) {
  unsigned u = __float_as_uint(f);
  u += 0x7FFFu + ((u >> 16) & 1u);          // round-to-nearest-even
  return (unsigned short)(u >> 16);
}
__device__ __forceinline__ unsigned pack_bf2(float lo, float hi) {
  return (unsigned)f2bf_bits(lo) | ((unsigned)f2bf_bits(hi) << 16);
}

// Fragment slot for K-index `kin` (0..31) within a 16x16x32 bf16 WMMA operand:
//   element i = (kin&7) | ((kin>>1)&8) ; lane gets +16 when (kin&8).
__device__ __forceinline__ int frag_elem(int kin) { return (kin & 7) | ((kin >> 1) & 8); }
__device__ __forceinline__ int frag_lanehi(int kin) { return (kin & 8) << 1; } // 0 or 16

// ------------------------------ utility ------------------------------------
__global__ void k_zero(float* __restrict__ p, long long n) {
  long long i = (long long)blockIdx.x * blockDim.x + threadIdx.x;
  if (i < n) p[i] = 0.0f;
}

__global__ void k_degree(const int* __restrict__ dst, float* __restrict__ deg, int E) {
  int e = blockIdx.x * blockDim.x + threadIdx.x;
  if (e < E) atomicAdd(&deg[dst[e]], 1.0f);
}

__global__ void k_dinv(const float* __restrict__ deg, float* __restrict__ dinv, int n) {
  int i = blockIdx.x * blockDim.x + threadIdx.x;
  if (i < n) dinv[i] = rsqrtf(deg[i] + 1.0f);   // self-loop degree
}

// ------------------------------ GEMM 1 -------------------------------------
// h[M,128] = X[M,512] @ W1[512,128]
// Block tile 64x128, 8 waves; wave w owns n-tile w and 4 m-subtiles. K tile 64.
// LDS is fragment-major: As[(mt*2+ks)*32+lane][16], Bs[(ks*8+nt)*32+lane][16].
__global__ __launch_bounds__(256) void k_gemm1(const float* __restrict__ X,
                                               const float* __restrict__ W,
                                               float* __restrict__ H, int Nn) {
  __shared__ __align__(32) unsigned short As[64 * 64];
  __shared__ __align__(32) unsigned short Bs[64 * 128];
  const int tid  = threadIdx.x;
  const int lane = tid & 31;
  const int wave = tid >> 5;             // n-tile id (0..7)
  const int m0   = blockIdx.x * 64;

  v8f acc[4] = {};

  for (int k0 = 0; k0 < IN_CH; k0 += 64) {
    // ---- A: 64 rows x 64 K, packed 2 bf16 per 32-bit LDS store ----
    for (int idx = tid; idx < 64 * 32; idx += 256) {
      int r = idx >> 5, k = (idx & 31) << 1;
      int row = m0 + r; if (row >= Nn) row = Nn - 1;
      float2 xv = *(const float2*)&X[(size_t)row * IN_CH + k0 + k];
      int kin = k & 31, ks = k >> 5;
      int ln = (r & 15) | frag_lanehi(kin);
      int addr = (((((r >> 4) << 1) + ks) << 5 | ln) << 4) + frag_elem(kin);
      *(unsigned*)&As[addr] = pack_bf2(xv.x, xv.y);
    }
    // ---- B: 64 K x 128 N, pairs along K ----
    for (int idx = tid; idx < 32 * 128; idx += 256) {
      int kp = idx >> 7, c = idx & 127;
      int k = kp << 1;
      float lo = W[(size_t)(k0 + k) * HID + c];
      float hi = W[(size_t)(k0 + k + 1) * HID + c];
      int kin = k & 31, ks = k >> 5;
      int ln = (c & 15) | frag_lanehi(kin);
      int addr = ((((ks << 3) + (c >> 4)) << 5 | ln) << 4) + frag_elem(kin);
      *(unsigned*)&Bs[addr] = pack_bf2(lo, hi);
    }
    __syncthreads();

#pragma unroll
    for (int ks = 0; ks < 2; ++ks) {
      v16bf b = *(const v16bf*)&Bs[(((ks << 3) + wave) << 5 | lane) << 4];
#pragma unroll
      for (int mt = 0; mt < 4; ++mt) {
        v16bf a = *(const v16bf*)&As[(((mt << 1) + ks) << 5 | lane) << 4];
        acc[mt] = __builtin_amdgcn_wmma_f32_16x16x32_bf16(false, a, false, b,
                                                          (short)0, acc[mt], false, false);
      }
    }
    __syncthreads();
  }

  const int mo = (lane & 16) ? 8 : 0;    // C/D layout: VGPR r -> M=r (+8 hi lanes)
  const int nn = lane & 15;
  const int n0 = wave * 16;
#pragma unroll
  for (int mt = 0; mt < 4; ++mt)
#pragma unroll
    for (int r = 0; r < 8; ++r) {
      int row = m0 + mt * 16 + mo + r;
      if (row < Nn) H[(size_t)row * HID + n0 + nn] = acc[mt][r];
    }
}

// ------------------------------ GEMM 2 -------------------------------------
// h2[M,64] = XR[M,128](bf16) @ W2[128,64]; block tile 64x64, K=128 in one pass.
__global__ __launch_bounds__(256) void k_gemm2(const unsigned short* __restrict__ XR,
                                               const float* __restrict__ W,
                                               float* __restrict__ H, int Nn) {
  __shared__ __align__(32) unsigned short As[64 * 128]; // [mt4][ks4][lane][16]
  __shared__ __align__(32) unsigned short Bs[128 * 64]; // [ks4][nt4][lane][16]
  const int tid   = threadIdx.x;
  const int lane  = tid & 31;
  const int wave  = tid >> 5;
  const int m0    = blockIdx.x * 64;
  const int nt    = wave & 3;
  const int mbase = wave >> 2;           // wave covers m-tiles mbase, mbase+2

  // ---- A: 64 rows x 128 K (already bf16; copy pairs) ----
  for (int idx = tid; idx < 64 * 64; idx += 256) {
    int r = idx >> 6, k = (idx & 63) << 1;
    int row = m0 + r; if (row >= Nn) row = Nn - 1;
    unsigned pv = *(const unsigned*)&XR[(size_t)row * HID + k];
    int kin = k & 31, ks = k >> 5;
    int ln = (r & 15) | frag_lanehi(kin);
    int addr = (((((r >> 4) << 2) + ks) << 5 | ln) << 4) + frag_elem(kin);
    *(unsigned*)&As[addr] = pv;
  }
  // ---- B: 128 K x 64 N ----
  for (int idx = tid; idx < 64 * 64; idx += 256) {
    int kp = idx >> 6, c = idx & 63;
    int k = kp << 1;
    float lo = W[(size_t)k * NCLS + c];
    float hi = W[(size_t)(k + 1) * NCLS + c];
    int kin = k & 31, ks = k >> 5;
    int ln = (c & 15) | frag_lanehi(kin);
    int addr = ((((ks << 2) + (c >> 4)) << 5 | ln) << 4) + frag_elem(kin);
    *(unsigned*)&Bs[addr] = pack_bf2(lo, hi);
  }
  __syncthreads();

  v8f acc[2] = {};
#pragma unroll
  for (int ks = 0; ks < 4; ++ks) {
    v16bf b = *(const v16bf*)&Bs[(((ks << 2) + nt) << 5 | lane) << 4];
#pragma unroll
    for (int t = 0; t < 2; ++t) {
      int mt = mbase + t * 2;
      v16bf a = *(const v16bf*)&As[(((mt << 2) + ks) << 5 | lane) << 4];
      acc[t] = __builtin_amdgcn_wmma_f32_16x16x32_bf16(false, a, false, b,
                                                       (short)0, acc[t], false, false);
    }
  }

  const int mo = (lane & 16) ? 8 : 0;
  const int nn = lane & 15;
  const int n0 = nt * 16;
#pragma unroll
  for (int t = 0; t < 2; ++t) {
    int mt = mbase + t * 2;
#pragma unroll
    for (int r = 0; r < 8; ++r) {
      int row = m0 + mt * 16 + mo + r;
      if (row < Nn) H[(size_t)row * NCLS + n0 + nn] = acc[t][r];
    }
  }
}

// ------------------------------ scatter ------------------------------------
// One lane per (edge, channel): gather h[src] (h+agg = 102MB << 192MB L2, so
// random traffic resolves on-chip), atomic scatter-add into agg[dst].
template <int LOGC>
__global__ void k_scatter(const float* __restrict__ H, const int* __restrict__ src,
                          const int* __restrict__ dst, const float* __restrict__ dinv,
                          float* __restrict__ agg, int E) {
  const int C = 1 << LOGC;
  long long gid = (long long)blockIdx.x * blockDim.x + threadIdx.x;
  int e = (int)(gid >> LOGC);
  int c = (int)(gid & (C - 1));
  if (e >= E) return;
  int s = src[e], d = dst[e];
  float w = dinv[s] * dinv[d];
  atomicAdd(&agg[(size_t)d * C + c], H[(size_t)s * C + c] * w);
}

// ------------------------------ finalize -----------------------------------
__global__ void k_finalize1(const float* __restrict__ agg, const float* __restrict__ H,
                            const float* __restrict__ dinv, const float* __restrict__ b,
                            unsigned short* __restrict__ XR, long long total) {
  long long gid = (long long)blockIdx.x * blockDim.x + threadIdx.x;
  if (gid >= total) return;
  int i = (int)(gid >> 7), c = (int)(gid & 127);
  float di = dinv[i];
  float v = agg[gid] + H[gid] * di * di + b[c];
  v = v > 0.0f ? v : 0.0f;                      // relu
  XR[gid] = f2bf_bits(v);
}

__global__ void k_finalize2(const float* __restrict__ agg, const float* __restrict__ H,
                            const float* __restrict__ dinv, const float* __restrict__ b,
                            float* __restrict__ out, long long total) {
  long long gid = (long long)blockIdx.x * blockDim.x + threadIdx.x;
  if (gid >= total) return;
  int i = (int)(gid >> 6), c = (int)(gid & 63);
  float di = dinv[i];
  out[gid] = agg[gid] + H[gid] * di * di + b[c];
}

// ------------------------------ launcher -----------------------------------
static inline unsigned blks(long long n, int b) { return (unsigned)((n + b - 1) / b); }

extern "C" void kernel_launch(void* const* d_in, const int* in_sizes, int n_in,
                              void* d_out, int out_size, void* d_ws, size_t ws_size,
                              hipStream_t stream) {
  const float* x    = (const float*)d_in[0];
  const int*   edge = (const int*)d_in[1];
  const float* W1   = (const float*)d_in[2];
  const float* b1   = (const float*)d_in[3];
  const float* W2   = (const float*)d_in[4];
  const float* b2   = (const float*)d_in[5];

  const int N = in_sizes[0] / IN_CH;     // 100000
  const int E = in_sizes[1] / 2;         // 1600000
  const int* srcp = edge;                // edge_index[0]
  const int* dstp = edge + E;            // edge_index[1]

  float* ws   = (float*)d_ws;
  float* deg  = ws;                                    // N
  float* dinv = ws + (size_t)N;                        // N
  float* h    = ws + (size_t)2 * N;                    // 128N (reused as 64N)
  float* agg  = ws + (size_t)(2 + HID) * N;            // 128N (reused as 64N)
  unsigned short* xr = (unsigned short*)(ws + (size_t)(2 + 2 * HID) * N); // 128N bf16

  float* outp = (float*)d_out;

  // ---- degree & normalization ----
  k_zero<<<blks(N, 256), 256, 0, stream>>>(deg, N);
  k_degree<<<blks(E, 256), 256, 0, stream>>>(dstp, deg, E);
  k_dinv<<<blks(N, 256), 256, 0, stream>>>(deg, dinv, N);

  // ---- layer 1 ----
  k_zero<<<blks((long long)N * HID, 256), 256, 0, stream>>>(agg, (long long)N * HID);
  k_gemm1<<<blks(N, 64), 256, 0, stream>>>(x, W1, h, N);
  k_scatter<7><<<blks((long long)E * HID, 256), 256, 0, stream>>>(h, srcp, dstp, dinv, agg, E);
  k_finalize1<<<blks((long long)N * HID, 256), 256, 0, stream>>>(agg, h, dinv, b1, xr,
                                                                 (long long)N * HID);

  // ---- layer 2 (reuse h/agg regions at 64 channels) ----
  k_zero<<<blks((long long)N * NCLS, 256), 256, 0, stream>>>(agg, (long long)N * NCLS);
  k_gemm2<<<blks(N, 64), 256, 0, stream>>>(xr, W2, h, N);
  k_scatter<6><<<blks((long long)E * NCLS, 256), 256, 0, stream>>>(h, srcp, dstp, dinv, agg, E);
  k_finalize2<<<blks((long long)N * NCLS, 256), 256, 0, stream>>>(agg, h, dinv, b2, outp,
                                                                  (long long)N * NCLS);
}
